// QMDPLayer_31284541784318
// MI455X (gfx1250) — compile-verified, matrix operand order checked by reading, not run on hardware
//
#include <hip/hip_runtime.h>

// ---------------------------------------------------------------------------
// QMDP scan layer for gfx1250 (MI455X).
// T=64, B=64, S=512, A=16, R=64, H=32.
//
// transition[k,i,j] = softmax_j( G[i,j] - 2*p[k,i]*p[k,j] ),  G = u u^T, p = u v_hat
// scan step:  s_next = (a (x) bel) [64 x 8192]  @  trans [8192 x 512]   <-- bf16 WMMA
//             b_post = softmax(log(s_next+eps) + logp_o[t])             -> alpha_b[t]
//             logits[h,k] = value[h,n,k,:] . b_post[n,:]; pi=softmax_k;
//             alpha_pi[t][n,k] = sum_h pi * tau_pdf[h]
//
// GEMM: dual-N-tile per wave (1 A load -> 2 v_wmma), split-K x16 with
// deterministic sequential reduction, operands pre-swizzled into WMMA lane order.
// ---------------------------------------------------------------------------

#define TT 64
#define BB 64
#define SS 512
#define AA 16
#define RR 64
#define HH 32
#define KTOT (AA * SS)       // 8192 contraction length
#define NKT (KTOT / 32)      // 256 K-tiles of 32
#define NNT (SS / 16)        // 32 N-tiles
#define NMT (BB / 16)        // 4  M-tiles
#define SPLITK 16
#define KT_PER (NKT / SPLITK)  // 16

typedef __attribute__((ext_vector_type(16))) __bf16 v16bf;
typedef __attribute__((ext_vector_type(8)))  float  v8f;

__device__ __forceinline__ unsigned short f2bf(float f) {
  union { float f; unsigned int u; } c; c.f = f;
  unsigned int u = c.u;
  unsigned int r = u + 0x7FFFu + ((u >> 16) & 1u);   // round-to-nearest-even
  return (unsigned short)(r >> 16);
}

__device__ __forceinline__ float wave_red_sum(float x) {
  #pragma unroll
  for (int m = 16; m >= 1; m >>= 1) x += __shfl_xor(x, m, 32);
  return x;
}

// ---------------- setup kernels -------------------------------------------

// normalize v rows -> vn [A,R]
__global__ void k_vn(const float* __restrict__ v, float* __restrict__ vn) {
  int k = blockIdx.x, tid = threadIdx.x;            // block = 64
  __shared__ float red[64];
  float x = v[k * RR + tid];
  red[tid] = x * x; __syncthreads();
  for (int s = 32; s > 0; s >>= 1) { if (tid < s) red[tid] += red[tid + s]; __syncthreads(); }
  vn[k * RR + tid] = x * rsqrtf(red[0]);
}

// uT[r][i] = u[i][r]
__global__ void k_uT(const float* __restrict__ u, float* __restrict__ uT) {
  int idx = blockIdx.x * blockDim.x + threadIdx.x;  // 32768
  int i = idx >> 6, r = idx & 63;
  uT[r * SS + i] = u[idx];
}

// p[k*S+i] = u[i,:] . vn[k,:]   (one wave per entry)
__global__ void k_p(const float* __restrict__ u, const float* __restrict__ vn,
                    float* __restrict__ p) {
  int gw = (blockIdx.x * blockDim.x + threadIdx.x) >> 5;   // 0..8191
  int lane = threadIdx.x & 31;
  int k = gw >> 9, i = gw & 511;
  float s = u[i * RR + lane] * vn[k * RR + lane]
          + u[i * RR + lane + 32] * vn[k * RR + lane + 32];
  s = wave_red_sum(s);
  if (lane == 0) p[gw] = s;
}

// G[i][j] = u[i,:] . u[j,:]  via uT for coalescing
__global__ void k_G(const float* __restrict__ u, const float* __restrict__ uT,
                    float* __restrict__ G) {
  int i = blockIdx.x, tid = threadIdx.x;            // block = 256
  __shared__ float ui[RR];
  if (tid < RR) ui[tid] = u[i * RR + tid];
  __syncthreads();
  for (int jj = tid; jj < SS; jj += 256) {
    float s = 0.f;
    #pragma unroll 8
    for (int r = 0; r < RR; ++r) s += ui[r] * uT[r * SS + jj];
    G[i * SS + jj] = s;
  }
}

// transition row softmax -> bf16, stored in WMMA B-operand tile order:
//   transW[((ktile*32 + ntile)*32 + lane)*16 + e]
__global__ void k_trans(const float* __restrict__ G, const float* __restrict__ p,
                        unsigned short* __restrict__ transW) {
  int kk = blockIdx.x;                  // 0..8191 = k*512+i
  int k = kk >> 9, i = kk & 511;
  int tid = threadIdx.x;                // 256
  __shared__ float red[256];
  float pki = p[k * SS + i];
  int j0 = tid, j1 = tid + 256;
  float w0 = G[i * SS + j0] - 2.f * pki * p[k * SS + j0];
  float w1 = G[i * SS + j1] - 2.f * pki * p[k * SS + j1];
  red[tid] = fmaxf(w0, w1); __syncthreads();
  for (int s = 128; s > 0; s >>= 1) { if (tid < s) red[tid] = fmaxf(red[tid], red[tid + s]); __syncthreads(); }
  float mx = red[0]; __syncthreads();
  float e0 = expf(w0 - mx), e1 = expf(w1 - mx);
  red[tid] = e0 + e1; __syncthreads();
  for (int s = 128; s > 0; s >>= 1) { if (tid < s) red[tid] += red[tid + s]; __syncthreads(); }
  float inv = 1.f / red[0];

  int ktile = kk >> 5, kin = kk & 31;
  int group = kin >> 3;
  int laneBase = 16 * (group & 1);
  int e = (kin & 7) + 8 * (group >> 1);
  {
    int ntile = j0 >> 4, lane = (j0 & 15) + laneBase;
    transW[(((ktile * NNT + ntile) * 32 + lane) << 4) + e] = f2bf(e0 * inv);
  }
  {
    int ntile = j1 >> 4, lane = (j1 & 15) + laneBase;
    transW[(((ktile * NNT + ntile) * 32 + lane) << 4) + e] = f2bf(e1 * inv);
  }
}

// normalized Poisson pmf over 1..H, rate = softplus(tau)
__global__ void k_taupdf(const float* __restrict__ tau, float* __restrict__ tpdf) {
  int k = threadIdx.x;                  // 32
  float rate = log1pf(expf(tau[0]));
  float kk = (float)(k + 1);
  float lp = kk * logf(rate) - rate - lgammaf(kk + 1.f);
  float pv = expf(lp);
  float s = wave_red_sum(pv);
  tpdf[k] = pv / s;
}

// alpha_a = softmax(logp_u) over A; one thread per (t,n) row
__global__ void k_alpha_a(const float* __restrict__ logp_u, float* __restrict__ a_all) {
  int row = blockIdx.x * blockDim.x + threadIdx.x;   // < 4096
  const float* src = logp_u + (size_t)row * AA;
  float mx = -1e30f;
  #pragma unroll
  for (int k = 0; k < AA; ++k) mx = fmaxf(mx, src[k]);
  float e[AA], s = 0.f;
  #pragma unroll
  for (int k = 0; k < AA; ++k) { e[k] = expf(src[k] - mx); s += e[k]; }
  float inv = 1.f / s;
  float* dst = a_all + (size_t)row * AA;
  #pragma unroll
  for (int k = 0; k < AA; ++k) dst[k] = e[k] * inv;
}

// ---------------- per-step kernels ----------------------------------------

// mW (WMMA A-operand tile order): mW[((mtile*NKT + ktile)*32 + lane)*16 + e]
//   value = a[t,n,k] * bel[n,i]   with kk = k*512+i the contraction index
__global__ void k_m(const float* __restrict__ a_t, const float* __restrict__ bel,
                    unsigned short* __restrict__ mW) {
  int idx = blockIdx.x * blockDim.x + threadIdx.x;   // < 524288
  int e = idx & 15;
  int lane = (idx >> 4) & 31;
  int ktile = (idx >> 9) & (NKT - 1);
  int mtile = idx >> 17;
  int m_in = lane & 15;
  int kin = (e & 7) + ((lane >> 4) << 3) + ((e >> 3) << 4);
  int kk = ktile * 32 + kin;
  int k = kk >> 9, i = kk & 511;
  int n = mtile * 16 + m_in;
  mW[idx] = f2bf(a_t[n * AA + k] * bel[n * SS + i]);
}

// split-K bf16 WMMA GEMM, dual N-tile per wave: s_part[sk][64][512] partials
__global__ void k_gemm(const unsigned short* __restrict__ mW,
                       const unsigned short* __restrict__ transW,
                       float* __restrict__ s_part) {
  int wave = threadIdx.x >> 5;
  int lane = threadIdx.x & 31;
  int tile = blockIdx.x * 8 + wave;          // 0..1023
  int mtile = tile >> 8;                     // 0..3
  int rem   = tile & 255;
  int np    = rem >> 4;                      // 0..15 (N-tile pair)
  int sk    = rem & 15;                      // 0..15 (split-K slab)
  int nt0   = np * 2;

  const unsigned short* aBase  = mW + ((size_t)mtile * NKT * 32 + lane) * 16;
  const unsigned short* b0Base = transW + ((size_t)nt0 * 32 + lane) * 16;
  const unsigned short* b1Base = transW + ((size_t)(nt0 + 1) * 32 + lane) * 16;
  int kt0 = sk * KT_PER;

  v8f acc0 = {0.f, 0.f, 0.f, 0.f, 0.f, 0.f, 0.f, 0.f};
  v8f acc1 = {0.f, 0.f, 0.f, 0.f, 0.f, 0.f, 0.f, 0.f};
  for (int kt = kt0; kt < kt0 + KT_PER; ++kt) {
    const v16bf av  = *reinterpret_cast<const v16bf*>(aBase  + (size_t)kt * 512);
    const v16bf bv0 = *reinterpret_cast<const v16bf*>(b0Base + (size_t)kt * 16384);
    const v16bf bv1 = *reinterpret_cast<const v16bf*>(b1Base + (size_t)kt * 16384);
    __builtin_prefetch(aBase  + (size_t)(kt + 1) * 512,   0, 1);
    __builtin_prefetch(b0Base + (size_t)(kt + 1) * 16384, 0, 1);
    __builtin_prefetch(b1Base + (size_t)(kt + 1) * 16384, 0, 1);
    acc0 = __builtin_amdgcn_wmma_f32_16x16x32_bf16(false, av, false, bv0,
                                                   (short)0, acc0, false, false);
    acc1 = __builtin_amdgcn_wmma_f32_16x16x32_bf16(false, av, false, bv1,
                                                   (short)0, acc1, false, false);
  }
  // C/D layout: lanes 0-15: M=r, N=lane; lanes 16-31: M=8+r, N=lane-16
  int col = lane & 15;
  int rbase = (lane >> 4) << 3;
  float* dst = s_part + (size_t)sk * BB * SS;
  int j0 = nt0 * 16 + col;
  #pragma unroll
  for (int r = 0; r < 8; ++r) {
    int n = mtile * 16 + rbase + r;
    dst[n * SS + j0]      = acc0[r];
    dst[n * SS + j0 + 16] = acc1[r];
  }
}

// reduce split-K partials -> s_next [64,512]
__global__ void k_reduce(const float* __restrict__ s_part, float* __restrict__ s_next) {
  int idx = blockIdx.x * blockDim.x + threadIdx.x;   // < 32768
  float s = 0.f;
  #pragma unroll
  for (int sk = 0; sk < SPLITK; ++sk) s += s_part[(size_t)sk * BB * SS + idx];
  s_next[idx] = s;
}

// b_post = softmax(log(s_next+eps) + logp_o[t]) per belief row -> alpha_b[t]
__global__ void k_bpost(const float* __restrict__ s_next,
                        const float* __restrict__ lo_t,
                        float* __restrict__ out_b_t) {
  int n = blockIdx.x, tid = threadIdx.x;    // block = 256
  __shared__ float red[256];
  int j0 = tid, j1 = tid + 256;
  float x0 = logf(s_next[n * SS + j0] + 1e-6f) + lo_t[n * SS + j0];
  float x1 = logf(s_next[n * SS + j1] + 1e-6f) + lo_t[n * SS + j1];
  red[tid] = fmaxf(x0, x1); __syncthreads();
  for (int s = 128; s > 0; s >>= 1) { if (tid < s) red[tid] = fmaxf(red[tid], red[tid + s]); __syncthreads(); }
  float mx = red[0]; __syncthreads();
  float e0 = expf(x0 - mx), e1 = expf(x1 - mx);
  red[tid] = e0 + e1; __syncthreads();
  for (int s = 128; s > 0; s >>= 1) { if (tid < s) red[tid] += red[tid + s]; __syncthreads(); }
  float inv = 1.f / red[0];
  out_b_t[n * SS + j0] = e0 * inv;
  out_b_t[n * SS + j1] = e1 * inv;
}

// plan: logits[h,k] = value[h,n,k,:].b_post ; pi=softmax_k ; out = sum_h pi*tpdf[h]
__global__ void k_plan(const float* __restrict__ out_b_t,
                       const float* __restrict__ value,
                       const float* __restrict__ tpdf,
                       float* __restrict__ out_pi_t) {
  int n = blockIdx.x;
  int tid = threadIdx.x;                    // block = 512 (16 waves)
  int wave = tid >> 5, lane = tid & 31;
  __shared__ float bp[SS];
  __shared__ float lg[HH * AA];             // 512
  bp[tid] = out_b_t[n * SS + tid];
  __syncthreads();
  #pragma unroll 1
  for (int q = 0; q < 32; ++q) {
    int pr = wave * 32 + q;                 // 0..511 -> (h,k)
    int h = pr >> 4, k = pr & 15;
    const float* vrow = value + (((size_t)h * BB + n) * AA + k) * SS;
    float s = 0.f;
    #pragma unroll 4
    for (int i = lane; i < SS; i += 32) s += bp[i] * vrow[i];
    s = wave_red_sum(s);
    if (lane == 0) lg[pr] = s;
  }
  __syncthreads();
  if (tid < HH) {                           // softmax over k for row h=tid
    int h = tid;
    float mx = -1e30f;
    #pragma unroll
    for (int k = 0; k < AA; ++k) mx = fmaxf(mx, lg[h * AA + k]);
    float e[AA], s = 0.f;
    #pragma unroll
    for (int k = 0; k < AA; ++k) { e[k] = expf(lg[h * AA + k] - mx); s += e[k]; }
    float scale = tpdf[h] / s;
    #pragma unroll
    for (int k = 0; k < AA; ++k) lg[h * AA + k] = e[k] * scale;
  }
  __syncthreads();
  if (tid < AA) {
    float s = 0.f;
    #pragma unroll
    for (int h = 0; h < HH; ++h) s += lg[h * AA + tid];
    out_pi_t[n * AA + tid] = s;
  }
}

// ---------------- launcher -------------------------------------------------

extern "C" void kernel_launch(void* const* d_in, const int* in_sizes, int n_in,
                              void* d_out, int out_size, void* d_ws, size_t ws_size,
                              hipStream_t stream) {
  const float* logp_o = (const float*)d_in[0];   // [T,B,S]
  const float* logp_u = (const float*)d_in[1];   // [T,B,A]
  const float* value  = (const float*)d_in[2];   // [H,B,A,S]
  const float* b0     = (const float*)d_in[3];   // [B,S]
  const float* u      = (const float*)d_in[4];   // [S,R]
  const float* v      = (const float*)d_in[5];   // [A,R]
  const float* tau    = (const float*)d_in[6];   // [1,1]

  float* out_b  = (float*)d_out;                         // [T,B,S]
  float* out_pi = (float*)d_out + (size_t)TT * BB * SS;  // [T,B,A]

  char* ws = (char*)d_ws;
  unsigned short* transW = (unsigned short*)ws; ws += (size_t)NKT * NNT * 32 * 16 * 2; // 8.39 MB
  unsigned short* mW     = (unsigned short*)ws; ws += (size_t)NMT * NKT * 32 * 16 * 2; // 1.05 MB
  float* G      = (float*)ws; ws += (size_t)SS * SS * 4;          // 1 MB
  float* uT     = (float*)ws; ws += (size_t)SS * RR * 4;          // 128 KB
  float* p      = (float*)ws; ws += (size_t)AA * SS * 4;          // 32 KB
  float* vn     = (float*)ws; ws += (size_t)AA * RR * 4;          // 4 KB
  float* a_all  = (float*)ws; ws += (size_t)TT * BB * AA * 4;     // 256 KB
  float* tpdf   = (float*)ws; ws += 256;
  float* s_part = (float*)ws; ws += (size_t)SPLITK * BB * SS * 4; // 2 MB
  float* s_next = (float*)ws; ws += (size_t)BB * SS * 4;          // 128 KB

  // ---- one-time setup ----
  k_vn     <<<AA, 64, 0, stream>>>(v, vn);
  k_uT     <<<(SS * RR) / 256, 256, 0, stream>>>(u, uT);
  k_p      <<<(AA * SS) / 4, 128, 0, stream>>>(u, vn, p);
  k_G      <<<SS, 256, 0, stream>>>(u, uT, G);
  k_trans  <<<AA * SS, 256, 0, stream>>>(G, p, transW);
  k_taupdf <<<1, 32, 0, stream>>>(tau, tpdf);
  k_alpha_a<<<(TT * BB) / 256, 256, 0, stream>>>(logp_u, a_all);

  // ---- sequential scan over T ----
  for (int t = 0; t < TT; ++t) {
    const float* bel = (t == 0) ? b0 : out_b + (size_t)(t - 1) * BB * SS;
    const float* a_t = a_all + (size_t)t * BB * AA;
    k_m      <<<2048, 256, 0, stream>>>(a_t, bel, mW);
    k_gemm   <<<128, 256, 0, stream>>>(mW, transW, s_part);
    k_reduce <<<128, 256, 0, stream>>>(s_part, s_next);
    k_bpost  <<<BB, 256, 0, stream>>>(s_next, logp_o + (size_t)t * BB * SS,
                                      out_b + (size_t)t * BB * SS);
    k_plan   <<<BB, 512, 0, stream>>>(out_b + (size_t)t * BB * SS, value, tpdf,
                                      out_pi + (size_t)t * BB * AA);
  }
}
